// MammothModa2Qwen2ForCausalLM_13434657702336
// MI455X (gfx1250) — matrix operand order repaired and checked by reading, not compiled, not run on hardware
//
#include <hip/hip_runtime.h>
#include <hip/hip_bf16.h>

// MoE-routed SwiGLU MLP for MI455X (gfx1250, wave32, WMMA).
// Route tokens to one expert (halves FLOPs vs reference), fp32 -> bf16 once,
// all GEMMs on v_wmma_f32_16x16x32_bf16 with fp32 accumulation.
// BK=64: two WMMA K-steps per LDS stage to amortize barriers.

typedef __bf16 bf16;
typedef bf16  v16bf __attribute__((ext_vector_type(16)));
typedef float v8f   __attribute__((ext_vector_type(8)));

#define T_TOK 16384
#define DDIM  2048
#define FDIM  5504
#define BM    128
#define BN    64
#define BK    64     // K elements staged per barrier phase (2 WMMA k-steps)

union FragBF { v16bf v; uint4 q[2]; };

// A fragment (16x32 bf16): lane l<16 -> row l, K chunks [0..7],[16..23];
// lane l>=16 -> row l-16, K chunks [8..15],[24..31]. LDS row stride = BK.
__device__ __forceinline__ v16bf load_a_frag(const bf16* base, int lane) {
  const int r  = lane & 15;
  const int k0 = (lane >> 4) * 8;
  const uint4* p = (const uint4*)(base + r * BK + k0);
  FragBF f; f.q[0] = p[0]; f.q[1] = p[2];   // second chunk at +16 elements
  return f.v;
}

// B fragment (32x16 bf16): lane l<16 -> col l, K chunks [0..7],[16..23];
// lane l>=16 -> col l-16, K chunks [8..15],[24..31]. LDS tile is column-major, row stride = BK.
__device__ __forceinline__ v16bf load_b_frag(const bf16* base, int lane) {
  const int n  = lane & 15;
  const int k0 = (lane >> 4) * 8;
  const uint4* p = (const uint4*)(base + n * BK + k0);
  FragBF f; f.q[0] = p[0]; f.q[1] = p[2];
  return f.v;
}

// ---------------------------------------------------------------- routing
__global__ void zero_counts_kernel(int* counts) {
  if (threadIdx.x < 2) counts[threadIdx.x] = 0;
}

__global__ void route_kernel(const unsigned char* __restrict__ mask,
                             int* __restrict__ lists, int* __restrict__ counts) {
  const int t = blockIdx.x * 256 + threadIdx.x;   // grid covers T exactly
  const int e = mask[t] ? 1 : 0;                  // 1 = gen, 0 = und
  const int idx = atomicAdd(&counts[e], 1);
  lists[e * T_TOK + idx] = t;
}

// ------------------------------------------------------- fp32 -> bf16 copies
struct BF4 { bf16 a, b, c, d; };   // 8 bytes

__global__ void cvt_bf16_kernel(const float* __restrict__ src,
                                bf16* __restrict__ dst, int n4) {
  const int i = blockIdx.x * 256 + threadIdx.x;
  if (i < n4) {
    const float4 v = ((const float4*)src)[i];
    BF4 o; o.a = (bf16)v.x; o.b = (bf16)v.y; o.c = (bf16)v.z; o.d = (bf16)v.w;
    ((BF4*)dst)[i] = o;
  }
}

// src: R x C fp32 row-major  ->  dst: C x R bf16 row-major (transposed)
__global__ void transpose_bf16_kernel(const float* __restrict__ src,
                                      bf16* __restrict__ dst, int R, int C) {
  __shared__ float tile[32][33];
  const int bx = blockIdx.x * 32;   // col base in src
  const int by = blockIdx.y * 32;   // row base in src
  const int tx = threadIdx.x;       // 0..31
  const int ty = threadIdx.y;       // 0..7
#pragma unroll
  for (int i = 0; i < 32; i += 8)
    tile[ty + i][tx] = src[(size_t)(by + ty + i) * C + (bx + tx)];
  __syncthreads();
#pragma unroll
  for (int i = 0; i < 32; i += 8)
    dst[(size_t)(bx + ty + i) * R + (by + tx)] = (bf16)tile[tx][ty + i];
}

// ------------------------------------------------------------------ GEMM 1
// H[r, f] = silu(X[tok] @ gate)[f] * (X[tok] @ up)[f], bf16 output.
// gen expert stores its compacted rows from the top of H (row T-1-r).
__global__ __launch_bounds__(256) void gemm1_kernel(
    const bf16* __restrict__ Xb,
    const bf16* __restrict__ gT0, const bf16* __restrict__ uT0,
    const bf16* __restrict__ gT1, const bf16* __restrict__ uT1,
    const int* __restrict__ lists, const int* __restrict__ counts,
    bf16* __restrict__ H) {
  __shared__ __align__(16) bf16 sA[BM * BK];   // 16 KB
  __shared__ __align__(16) bf16 sG[BN * BK];   // 8 KB
  __shared__ __align__(16) bf16 sU[BN * BK];   // 8 KB

  const int e    = blockIdx.z;
  const int cnt  = counts[e];
  const int row0 = blockIdx.y * BM;
  if (row0 >= cnt) return;
  const int col0 = blockIdx.x * BN;

  const int tid  = threadIdx.x;
  const int lane = tid & 31;
  const int wave = tid >> 5;
  const int wm   = wave >> 1;   // 0..3 -> 32 rows each
  const int wn   = wave & 1;    // 0..1 -> 32 cols each

  const int*  list = lists + e * T_TOK;
  const bf16* gT   = e ? gT1 : gT0;
  const bf16* uT   = e ? uT1 : uT0;

  // staging: 16B chunk per thread; 8 chunks per BK-row
  const int arb = tid >> 3;            // base row 0..31
  const int ac  = (tid & 7) * 8;       // bf16 offset of 16B chunk
  const int cm1 = cnt - 1;
  size_t aoff[4];
#pragma unroll
  for (int t = 0; t < 4; ++t)
    aoff[t] = (size_t)list[min(row0 + arb + t * 32, cm1)] * DDIM + ac;
  size_t boff[2];
#pragma unroll
  for (int t = 0; t < 2; ++t)
    boff[t] = (size_t)(col0 + arb + t * 32) * DDIM + ac;

  v8f cg[2][2] = {};
  v8f cu[2][2] = {};

  for (int kt = 0; kt < DDIM; kt += BK) {
#pragma unroll
    for (int t = 0; t < 4; ++t)
      *(uint4*)&sA[(arb + t * 32) * BK + ac] = *(const uint4*)&Xb[aoff[t] + kt];
#pragma unroll
    for (int t = 0; t < 2; ++t) {
      *(uint4*)&sG[(arb + t * 32) * BK + ac] = *(const uint4*)&gT[boff[t] + kt];
      *(uint4*)&sU[(arb + t * 32) * BK + ac] = *(const uint4*)&uT[boff[t] + kt];
    }
    if (kt + BK < DDIM) {   // prefetch next K tile (global_prefetch_b8)
      __builtin_prefetch(&Xb[aoff[0] + kt + BK], 0, 1);
      __builtin_prefetch(&gT[boff[0] + kt + BK], 0, 1);
      __builtin_prefetch(&uT[boff[0] + kt + BK], 0, 1);
    }
    __syncthreads();

#pragma unroll
    for (int ks = 0; ks < 2; ++ks) {
      const int ko = ks * 32;
      const v16bf a0 = load_a_frag(&sA[(wm * 32 +  0) * BK + ko], lane);
      const v16bf a1 = load_a_frag(&sA[(wm * 32 + 16) * BK + ko], lane);
      const v16bf g0 = load_b_frag(&sG[(wn * 32 +  0) * BK + ko], lane);
      const v16bf g1 = load_b_frag(&sG[(wn * 32 + 16) * BK + ko], lane);
      const v16bf u0 = load_b_frag(&sU[(wn * 32 +  0) * BK + ko], lane);
      const v16bf u1 = load_b_frag(&sU[(wn * 32 + 16) * BK + ko], lane);

      cg[0][0] = __builtin_amdgcn_wmma_f32_16x16x32_bf16(false, a0, false, g0, (short)0, cg[0][0], false, false);
      cg[0][1] = __builtin_amdgcn_wmma_f32_16x16x32_bf16(false, a0, false, g1, (short)0, cg[0][1], false, false);
      cg[1][0] = __builtin_amdgcn_wmma_f32_16x16x32_bf16(false, a1, false, g0, (short)0, cg[1][0], false, false);
      cg[1][1] = __builtin_amdgcn_wmma_f32_16x16x32_bf16(false, a1, false, g1, (short)0, cg[1][1], false, false);
      cu[0][0] = __builtin_amdgcn_wmma_f32_16x16x32_bf16(false, a0, false, u0, (short)0, cu[0][0], false, false);
      cu[0][1] = __builtin_amdgcn_wmma_f32_16x16x32_bf16(false, a0, false, u1, (short)0, cu[0][1], false, false);
      cu[1][0] = __builtin_amdgcn_wmma_f32_16x16x32_bf16(false, a1, false, u0, (short)0, cu[1][0], false, false);
      cu[1][1] = __builtin_amdgcn_wmma_f32_16x16x32_bf16(false, a1, false, u1, (short)0, cu[1][1], false, false);
    }
    __syncthreads();
  }

  // Fused SwiGLU epilogue -> bf16 H. C layout: n = lane&15, m = (lane>>4)*8+i.
  const int n_l    = lane & 15;
  const int m_base = (lane >> 4) * 8;
#pragma unroll
  for (int mi = 0; mi < 2; ++mi)
#pragma unroll
    for (int ni = 0; ni < 2; ++ni)
#pragma unroll
      for (int i = 0; i < 8; ++i) {
        const int rr = row0 + wm * 32 + mi * 16 + m_base + i;
        if (rr < cnt) {
          const float g = cg[mi][ni][i];
          const float u = cu[mi][ni][i];
          const float h = (g / (1.0f + __expf(-g))) * u;
          const int hrow = e ? (T_TOK - 1 - rr) : rr;
          const int cc   = col0 + wn * 32 + ni * 16 + n_l;
          H[(size_t)hrow * FDIM + cc] = (bf16)h;
        }
      }
}

// ------------------------------------------------------------------ GEMM 2
// out[tok, :] = H[r, :] @ down   (fp32 output, scattered by token list)
__global__ __launch_bounds__(256) void gemm2_kernel(
    const bf16* __restrict__ H,
    const bf16* __restrict__ dT0, const bf16* __restrict__ dT1,
    const int* __restrict__ lists, const int* __restrict__ counts,
    float* __restrict__ out) {
  __shared__ __align__(16) bf16 sA[BM * BK];   // 16 KB
  __shared__ __align__(16) bf16 sB[BN * BK];   // 8 KB

  const int e    = blockIdx.z;
  const int cnt  = counts[e];
  const int row0 = blockIdx.y * BM;
  if (row0 >= cnt) return;
  const int col0 = blockIdx.x * BN;

  const int tid  = threadIdx.x;
  const int lane = tid & 31;
  const int wave = tid >> 5;
  const int wm   = wave >> 1;
  const int wn   = wave & 1;

  const int*  list = lists + e * T_TOK;
  const bf16* dT   = e ? dT1 : dT0;

  const int arb = tid >> 3;
  const int ac  = (tid & 7) * 8;
  size_t aoff[4];
#pragma unroll
  for (int t = 0; t < 4; ++t) {
    const int rr = row0 + arb + t * 32;
    const int hr = e ? (T_TOK - 1 - rr) : rr;
    aoff[t] = (size_t)hr * FDIM + ac;
  }
  size_t boff[2];
#pragma unroll
  for (int t = 0; t < 2; ++t)
    boff[t] = (size_t)(col0 + arb + t * 32) * FDIM + ac;

  v8f c[2][2] = {};

  for (int kt = 0; kt < FDIM; kt += BK) {
#pragma unroll
    for (int t = 0; t < 4; ++t)
      *(uint4*)&sA[(arb + t * 32) * BK + ac] = *(const uint4*)&H[aoff[t] + kt];
#pragma unroll
    for (int t = 0; t < 2; ++t)
      *(uint4*)&sB[(arb + t * 32) * BK + ac] = *(const uint4*)&dT[boff[t] + kt];
    if (kt + BK < FDIM) {
      __builtin_prefetch(&H[aoff[0] + kt + BK], 0, 1);
      __builtin_prefetch(&dT[boff[0] + kt + BK], 0, 1);
    }
    __syncthreads();

#pragma unroll
    for (int ks = 0; ks < 2; ++ks) {
      const int ko = ks * 32;
      const v16bf a0 = load_a_frag(&sA[(wm * 32 +  0) * BK + ko], lane);
      const v16bf a1 = load_a_frag(&sA[(wm * 32 + 16) * BK + ko], lane);
      const v16bf b0 = load_b_frag(&sB[(wn * 32 +  0) * BK + ko], lane);
      const v16bf b1 = load_b_frag(&sB[(wn * 32 + 16) * BK + ko], lane);

      c[0][0] = __builtin_amdgcn_wmma_f32_16x16x32_bf16(false, a0, false, b0, (short)0, c[0][0], false, false);
      c[0][1] = __builtin_amdgcn_wmma_f32_16x16x32_bf16(false, a0, false, b1, (short)0, c[0][1], false, false);
      c[1][0] = __builtin_amdgcn_wmma_f32_16x16x32_bf16(false, a1, false, b0, (short)0, c[1][0], false, false);
      c[1][1] = __builtin_amdgcn_wmma_f32_16x16x32_bf16(false, a1, false, b1, (short)0, c[1][1], false, false);
    }
    __syncthreads();
  }

  const int n_l    = lane & 15;
  const int m_base = (lane >> 4) * 8;
#pragma unroll
  for (int mi = 0; mi < 2; ++mi)
#pragma unroll
    for (int ni = 0; ni < 2; ++ni)
#pragma unroll
      for (int i = 0; i < 8; ++i) {
        const int rr = row0 + wm * 32 + mi * 16 + m_base + i;
        if (rr < cnt) {
          const int tok = list[rr];
          const int cc  = col0 + wn * 32 + ni * 16 + n_l;
          out[(size_t)tok * DDIM + cc] = c[mi][ni][i];
        }
      }
}

// ------------------------------------------------------------------ launch
extern "C" void kernel_launch(void* const* d_in, const int* in_sizes, int n_in,
                              void* d_out, int out_size, void* d_ws, size_t ws_size,
                              hipStream_t stream) {
  const float* hidden   = (const float*)d_in[0];
  const float* und_gate = (const float*)d_in[1];
  const float* und_up   = (const float*)d_in[2];
  const float* und_down = (const float*)d_in[3];
  const float* gen_gate = (const float*)d_in[4];
  const float* gen_up   = (const float*)d_in[5];
  const float* gen_down = (const float*)d_in[6];
  const unsigned char* mask = (const unsigned char*)d_in[7];  // jax bool = 1 byte
  float* out = (float*)d_out;

  // workspace layout (bytes)
  char* ws = (char*)d_ws;
  const size_t OFF_LISTS = 256;                                   // 2*T ints
  const size_t OFF_X     = OFF_LISTS + (size_t)2 * T_TOK * 4;     // T*D bf16
  const size_t OFF_W     = OFF_X + (size_t)T_TOK * DDIM * 2;      // 6 transposed weights
  const size_t WSZ       = (size_t)DDIM * FDIM * 2;               // one bf16 weight
  const size_t OFF_H     = OFF_W + 6 * WSZ;                       // T*F bf16
  int*  counts = (int*)ws;
  int*  lists  = (int*)(ws + OFF_LISTS);
  bf16* Xb     = (bf16*)(ws + OFF_X);
  bf16* wgT0   = (bf16*)(ws + OFF_W + 0 * WSZ);
  bf16* wuT0   = (bf16*)(ws + OFF_W + 1 * WSZ);
  bf16* wdT0   = (bf16*)(ws + OFF_W + 2 * WSZ);
  bf16* wgT1   = (bf16*)(ws + OFF_W + 3 * WSZ);
  bf16* wuT1   = (bf16*)(ws + OFF_W + 4 * WSZ);
  bf16* wdT1   = (bf16*)(ws + OFF_W + 5 * WSZ);
  bf16* H      = (bf16*)(ws + OFF_H);

  // 1) token routing
  zero_counts_kernel<<<1, 32, 0, stream>>>(counts);
  route_kernel<<<T_TOK / 256, 256, 0, stream>>>(mask, lists, counts);

  // 2) precision conversion (+ weight transposes so B tiles are K-contiguous)
  {
    const int n4 = T_TOK * DDIM / 4;
    cvt_bf16_kernel<<<(n4 + 255) / 256, 256, 0, stream>>>(hidden, Xb, n4);
    dim3 blk(32, 8);
    dim3 gDxF(FDIM / 32, DDIM / 32);   // src D x F -> dst F x D
    dim3 gFxD(DDIM / 32, FDIM / 32);   // src F x D -> dst D x F
    transpose_bf16_kernel<<<gDxF, blk, 0, stream>>>(und_gate, wgT0, DDIM, FDIM);
    transpose_bf16_kernel<<<gDxF, blk, 0, stream>>>(und_up,   wuT0, DDIM, FDIM);
    transpose_bf16_kernel<<<gFxD, blk, 0, stream>>>(und_down, wdT0, FDIM, DDIM);
    transpose_bf16_kernel<<<gDxF, blk, 0, stream>>>(gen_gate, wgT1, DDIM, FDIM);
    transpose_bf16_kernel<<<gDxF, blk, 0, stream>>>(gen_up,   wuT1, DDIM, FDIM);
    transpose_bf16_kernel<<<gFxD, blk, 0, stream>>>(gen_down, wdT1, FDIM, DDIM);
  }

  // 3) gate/up GEMM + fused SwiGLU  (grid: F/BN x T/BM x experts)
  gemm1_kernel<<<dim3(FDIM / BN, T_TOK / BM, 2), 256, 0, stream>>>(
      Xb, wgT0, wuT0, wgT1, wuT1, lists, counts, H);

  // 4) down GEMM + scatter to output  (grid: D/BN x T/BM x experts)
  gemm2_kernel<<<dim3(DDIM / BN, T_TOK / BM, 2), 256, 0, stream>>>(
      H, wdT0, wdT1, lists, counts, out);
}
